// MultiHeadAttention_69509750718724
// MI455X (gfx1250) — compile-verified, matrix operand order checked by reading, not run on hardware
//
#include <hip/hip_runtime.h>

// ---------------------------------------------------------------------------
// MultiHeadAttention for MI455X (gfx1250): bf16 WMMA pipeline, fp32 accumulate,
// double-buffered async global->LDS copies (ASYNCcnt path), hw bf16 packing.
// B=4 T=2048 C=1024 H=16 HD=64
// ---------------------------------------------------------------------------

typedef __attribute__((ext_vector_type(16))) __bf16 v16bf;
typedef __attribute__((ext_vector_type(8)))  float  v8f;

union FragBF {
    v16bf v;
    unsigned int u[8];
};

#define B_  4
#define T_  2048
#define C_  1024
#define H_  16
#define HD_ 64

// hardware packed f32x2 -> bf16x2 (RNE), CDNA4/5 VOP3
__device__ __forceinline__ unsigned cvt2bf(float a, float b) {
    unsigned r;
    asm("v_cvt_pk_bf16_f32 %0, %1, %2" : "=v"(r) : "v"(a), "v"(b));
    return r;
}
__device__ __forceinline__ unsigned short f2bf(float f) {
    return (unsigned short)(cvt2bf(f, 0.0f) & 0xffffu);
}

// K-dim word index inside a 16-bit WMMA fragment (ISA 7.12.2):
// K = 16*(v>=4) + 8*half + 2*(v&3) + p  ->  u32 word = 8*(v>=4) + 4*half + (v&3)
__device__ __forceinline__ int frag_word(int v, int half) {
    return (v & 3) + 4 * half + ((v >> 2) << 3);
}

// flat address of a __shared__ object: low 32 bits == LDS byte offset (ISA 10.2)
__device__ __forceinline__ unsigned lds_off(const void* p) {
    return (unsigned)(unsigned long long)p;
}

// CDNA5 async copy: 16 contiguous bytes global -> 16 contiguous bytes LDS.
__device__ __forceinline__ void async_ld_b128(unsigned lds_addr, const void* gaddr) {
    asm volatile("global_load_async_to_lds_b128 %0, %1, off"
                 :
                 : "v"(lds_addr), "v"((unsigned long long)gaddr)
                 : "memory");
}
__device__ __forceinline__ void wait_async0() {
    asm volatile("s_wait_asynccnt 0x0" ::: "memory");
}

// ---------------------------------------------------------------------------
// Kernel 1a: fp32 -> bf16 convert, 2 elements/thread via v_cvt_pk_bf16_f32
// ---------------------------------------------------------------------------
__global__ void k_cvt_bf16(const float* __restrict__ src,
                           unsigned int* __restrict__ dst, int n2) {
    int i = blockIdx.x * blockDim.x + threadIdx.x;   // over n/2
    if (i < n2) dst[i] = cvt2bf(src[2 * i], src[2 * i + 1]);
}

// ---------------------------------------------------------------------------
// Kernel 1b: repack Wq/Wk/Wv (H,C,HD) -> fused (3, C, H*HD) bf16, 2 elems/thr
// ---------------------------------------------------------------------------
__global__ void k_pack_qkv(const float* __restrict__ Wq,
                           const float* __restrict__ Wk,
                           const float* __restrict__ Wv,
                           unsigned int* __restrict__ dst) {
    int i = blockIdx.x * blockDim.x + threadIdx.x;   // over C_*C_/2
    if (i >= C_ * C_ / 2) return;
    int e = 2 * i;
    int c = e >> 10, n = e & 1023;
    int h = n >> 6, d = n & 63;                       // d even
    size_t si = ((size_t)h * C_ + c) * HD_ + d;
    dst[i]                   = cvt2bf(Wq[si], Wq[si + 1]);
    dst[C_ * C_ / 2 + i]     = cvt2bf(Wk[si], Wk[si + 1]);
    dst[2 * (C_ * C_ / 2) + i] = cvt2bf(Wv[si], Wv[si + 1]);
}

// ---------------------------------------------------------------------------
// GEMM helpers: block tile 128(M) x 128(N), 8 waves (4M x 2N),
// wave tile 32x64 => 8 WMMAs per 32-deep K step, double-buffered LDS.
// A tile: async b128, stride 20 words (16B aligned, 16-bank-distinct rows).
// B tile: manual transposed store [n][k] (stride 34 bf16).
// ---------------------------------------------------------------------------
__device__ __forceinline__ void gemm_load_tile(
    const unsigned short* __restrict__ Au, const unsigned int* __restrict__ Wu,
    unsigned ldsA_base, unsigned short* ldsBt,
    int tid, int t0, int n0, int kk) {
#pragma unroll
    for (int jj = 0; jj < 2; jj++) {            // 512 b128 chunks of A
        int id = tid + jj * 256;
        int r = id >> 2, c = id & 3;
        unsigned la = ldsA_base + (unsigned)(r * 20 + c * 4) * 4u;
        const char* ga = (const char*)Au +
            ((size_t)(t0 + r) * 1024 + (size_t)kk) * 2 + (size_t)c * 16;
        async_ld_b128(la, ga);
    }
#pragma unroll
    for (int jj = 0; jj < 8; jj++) {            // 2048 u32 of B, transpose
        int id = tid + jj * 256;
        int r = id >> 6, w = id & 63;           // r = k row, w = u32 col
        unsigned int val = Wu[(size_t)(kk + r) * 512 + (n0 >> 1) + w];
        ldsBt[(2 * w + 0) * 34 + r] = (unsigned short)(val & 0xffffu);
        ldsBt[(2 * w + 1) * 34 + r] = (unsigned short)(val >> 16);
    }
}

__device__ __forceinline__ void gemm_compute(
    const unsigned int* ldsA, const unsigned short* ldsBt,
    int wm, int wn, int lrow, int half, v8f (&acc)[2][4]) {
    FragBF a[2];
#pragma unroll
    for (int mi = 0; mi < 2; mi++) {
        int row = wm * 32 + mi * 16 + lrow;
#pragma unroll
        for (int v = 0; v < 8; v++)
            a[mi].u[v] = ldsA[row * 20 + frag_word(v, half)];
    }
    const unsigned int* Btu = (const unsigned int*)ldsBt;
#pragma unroll
    for (int j = 0; j < 4; j++) {
        FragBF bf;
        int col = wn * 64 + j * 16 + lrow;
#pragma unroll
        for (int v = 0; v < 8; v++)
            bf.u[v] = Btu[col * 17 + frag_word(v, half)];
#pragma unroll
        for (int mi = 0; mi < 2; mi++)
            acc[mi][j] = __builtin_amdgcn_wmma_f32_16x16x32_bf16(
                false, a[mi].v, false, bf.v, (short)0, acc[mi][j], false, false);
    }
}

// ---------------------------------------------------------------------------
// Kernel 2: QKV projection GEMM (8192 x 1024)*(1024 x 1024) per z in {q,k,v}
// outputs scattered to (B,H,T,HD) bf16
// ---------------------------------------------------------------------------
__launch_bounds__(256)
__global__ void k_gemm_qkv(const unsigned short* __restrict__ Xb,
                           const unsigned short* __restrict__ Wb,  // 3 x C x 1024
                           unsigned short* __restrict__ Qb,
                           unsigned short* __restrict__ Kb,
                           unsigned short* __restrict__ Vb) {
    __shared__ unsigned int   ldsA[2][128 * 20];
    __shared__ unsigned short ldsBt[2][128 * 34];

    const int tid  = threadIdx.x;
    const int wave = tid >> 5;
    const int lane = tid & 31;
    const int lrow = lane & 15;
    const int half = lane >> 4;
    const int wm = wave & 3, wn = wave >> 2;
    const int t0 = blockIdx.x * 128;
    const int n0 = blockIdx.y * 128;
    const int m  = blockIdx.z;

    const unsigned short* Au = Xb;
    const unsigned int*   Wu = (const unsigned int*)(Wb + (size_t)m * C_ * 1024);
    const unsigned ldsA_base0 = lds_off(&ldsA[0][0]);
    const unsigned ldsA_base1 = lds_off(&ldsA[1][0]);

    v8f acc[2][4];
#pragma unroll
    for (int mi = 0; mi < 2; mi++)
#pragma unroll
        for (int j = 0; j < 4; j++)
#pragma unroll
            for (int e = 0; e < 8; e++) acc[mi][j][e] = 0.0f;

    gemm_load_tile(Au, Wu, ldsA_base0, ldsBt[0], tid, t0, n0, 0);
    wait_async0();
    __syncthreads();
    for (int kt = 0; kt < 32; kt++) {
        int buf = kt & 1;
        if (kt + 1 < 32)
            gemm_load_tile(Au, Wu, buf ? ldsA_base0 : ldsA_base1,
                           ldsBt[buf ^ 1], tid, t0, n0, (kt + 1) * 32);
        gemm_compute(ldsA[buf], ldsBt[buf], wm, wn, lrow, half, acc);
        wait_async0();
        __syncthreads();
    }

    unsigned short* dst = (m == 0) ? Qb : ((m == 1) ? Kb : Vb);
#pragma unroll
    for (int mi = 0; mi < 2; mi++)
#pragma unroll
        for (int j = 0; j < 4; j++) {
            int n = n0 + wn * 64 + j * 16 + lrow;
            int h = n >> 6, d = n & 63;
#pragma unroll
            for (int r = 0; r < 8; r += 2) {
                int tg = t0 + wm * 32 + mi * 16 + r + 8 * half;
                int b  = tg >> 11, t = tg & 2047;
                unsigned pk = cvt2bf(acc[mi][j][r], acc[mi][j][r + 1]);
                size_t base = (((size_t)(b * H_ + h)) * T_ + t) * HD_ + d;
                dst[base]       = (unsigned short)(pk & 0xffffu);
                dst[base + HD_] = (unsigned short)(pk >> 16);
            }
        }
}

// ---------------------------------------------------------------------------
// Kernel 3: flash attention. grid (T/64, H, B), 128 threads = 4 waves.
// wave w owns 16 query rows; 32-key blocks double-buffered through LDS.
// ---------------------------------------------------------------------------
__launch_bounds__(128)
__global__ void k_attn(const unsigned short* __restrict__ Qb,
                       const unsigned short* __restrict__ Kb,
                       const unsigned short* __restrict__ Vb,
                       unsigned short* __restrict__ attOut) {
    __shared__ unsigned int   ldsK[2][32 * 36];    // [key][d] stride 36 words
    __shared__ unsigned short ldsVt[2][64 * 34];   // [d][key] (+2 pad)
    __shared__ unsigned short ldsP[4 * 16 * 34];   // per-wave P staging

    const int tid  = threadIdx.x;
    const int wave = tid >> 5;
    const int lane = tid & 31;
    const int lrow = lane & 15;
    const int half = lane >> 4;
    const int blk = blockIdx.x, h = blockIdx.y, b = blockIdx.z;
    const size_t bh = (size_t)(b * H_ + h);

    const unsigned short* Kg = Kb + bh * T_ * HD_;
    const unsigned int* Qu = (const unsigned int*)(Qb + bh * T_ * HD_);
    const unsigned int* Vu = (const unsigned int*)(Vb + bh * T_ * HD_);
    const unsigned ldsK_base0 = lds_off(&ldsK[0][0]);
    const unsigned ldsK_base1 = lds_off(&ldsK[1][0]);

    const int q0 = blk * 64 + wave * 16;

    // Q fragments (two k=32 chunks over HD=64), loop invariant
    FragBF aQ[2];
#pragma unroll
    for (int c = 0; c < 2; c++)
#pragma unroll
        for (int v = 0; v < 8; v++)
            aQ[c].u[v] = Qu[(size_t)(q0 + lrow) * 32 + frag_word(v, half) + c * 16];

    v8f O[4];
    float mrow[8], lsum[8];
#pragma unroll
    for (int j = 0; j < 4; j++)
#pragma unroll
        for (int e = 0; e < 8; e++) O[j][e] = 0.0f;
#pragma unroll
    for (int r = 0; r < 8; r++) { mrow[r] = -1e30f; lsum[r] = 0.0f; }

    // tile loader: K async (layout already matches B-frag), V manual transpose
    auto load_kv = [&](int dstbuf, int kk) {
        unsigned kbase = dstbuf ? ldsK_base1 : ldsK_base0;
#pragma unroll
        for (int jj = 0; jj < 2; jj++) {           // 256 b128 chunks of K
            int id = tid + jj * 128;
            int r = id >> 3, c = id & 7;
            unsigned la = kbase + (unsigned)(r * 36 + c * 4) * 4u;
            const char* ga = (const char*)Kg +
                ((size_t)(kk + r) * 64) * 2 + (size_t)c * 16;
            async_ld_b128(la, ga);
        }
        unsigned short* vt = ldsVt[dstbuf];
#pragma unroll
        for (int jj = 0; jj < 8; jj++) {           // 1024 u32 of V, transpose
            int idx = tid + jj * 128;
            int r = idx >> 5, w = idx & 31;
            unsigned int val = Vu[(size_t)(kk + r) * 32 + w];
            vt[(2 * w + 0) * 34 + r] = (unsigned short)(val & 0xffffu);
            vt[(2 * w + 1) * 34 + r] = (unsigned short)(val >> 16);
        }
    };

    const int nkb = 2 * blk + 2;     // causal bound for the whole block
    load_kv(0, 0);
    wait_async0();
    __syncthreads();

    for (int kb = 0; kb < nkb; kb++) {
        const int kk = kb * 32;
        const int buf = kb & 1;
        if (kb + 1 < nkb) load_kv(buf ^ 1, kk + 32);

        // S = Q * K^T : two 16-key N tiles, each 2 WMMAs over d chunks
        const unsigned int* Kt = ldsK[buf];
        v8f s[2];
#pragma unroll
        for (int jn = 0; jn < 2; jn++) {
#pragma unroll
            for (int e = 0; e < 8; e++) s[jn][e] = 0.0f;
            int key = jn * 16 + lrow;
#pragma unroll
            for (int c = 0; c < 2; c++) {
                FragBF bK;
#pragma unroll
                for (int v = 0; v < 8; v++)
                    bK.u[v] = Kt[key * 36 + frag_word(v, half) + c * 16];
                s[jn] = __builtin_amdgcn_wmma_f32_16x16x32_bf16(
                    false, aQ[c].v, false, bK.v, (short)0, s[jn], false, false);
            }
        }

        // online softmax (row = r + 8*half lives across the lane's 16-lane half)
        unsigned short* Pst = ldsP + wave * 16 * 34;
        const float scale = 0.125f;   // 1/sqrt(64)
#pragma unroll
        for (int r = 0; r < 8; r++) {
            int q  = q0 + r + 8 * half;
            float s0 = s[0][r] * scale;
            float s1 = s[1][r] * scale;
            if (kk + lrow > q)      s0 = -1e30f;
            if (kk + 16 + lrow > q) s1 = -1e30f;
            float mx = fmaxf(s0, s1);
#pragma unroll
            for (int d = 1; d < 16; d <<= 1) mx = fmaxf(mx, __shfl_xor(mx, d));
            float mnew = fmaxf(mrow[r], mx);
            float corr = __expf(mrow[r] - mnew);
            float p0 = __expf(s0 - mnew);
            float p1 = __expf(s1 - mnew);
            float ps = p0 + p1;
#pragma unroll
            for (int d = 1; d < 16; d <<= 1) ps += __shfl_xor(ps, d);
            lsum[r] = lsum[r] * corr + ps;
            mrow[r] = mnew;
#pragma unroll
            for (int j = 0; j < 4; j++) O[j][r] *= corr;
            int prow = r + 8 * half;
            unsigned pk = cvt2bf(p0, p1);
            Pst[prow * 34 + lrow]      = (unsigned short)(pk & 0xffffu);
            Pst[prow * 34 + 16 + lrow] = (unsigned short)(pk >> 16);
        }
        // LDS ops of one wave are in-order; just fence compiler reordering
        __builtin_amdgcn_wave_barrier();

        // O += P * V  (contraction over 32 keys = one WMMA per 16-d tile)
        const unsigned int* Pu = (const unsigned int*)Pst;
        FragBF aP;
#pragma unroll
        for (int v = 0; v < 8; v++)
            aP.u[v] = Pu[lrow * 17 + frag_word(v, half)];
        const unsigned int* Vt = (const unsigned int*)ldsVt[buf];
#pragma unroll
        for (int j = 0; j < 4; j++) {
            FragBF bV;
            int col = j * 16 + lrow;
#pragma unroll
            for (int v = 0; v < 8; v++)
                bV.u[v] = Vt[col * 17 + frag_word(v, half)];
            O[j] = __builtin_amdgcn_wmma_f32_16x16x32_bf16(
                false, aP.v, false, bV.v, (short)0, O[j], false, false);
        }

        wait_async0();         // next-tile async K arrived in other buffer
        __syncthreads();       // publish next tile; retire this buffer
    }

    // normalize + write (B,T,C) bf16, head-concat order
#pragma unroll
    for (int j = 0; j < 4; j++) {
#pragma unroll
        for (int r = 0; r < 8; r += 2) {
            int t = q0 + r + 8 * half;
            unsigned pk = cvt2bf(O[j][r] / lsum[r], O[j][r + 1] / lsum[r + 1]);
            size_t base = ((size_t)(b * T_ + t)) * C_ + h * HD_ + j * 16 + lrow;
            attOut[base]      = (unsigned short)(pk & 0xffffu);
            attOut[base + C_] = (unsigned short)(pk >> 16);
        }
    }
}

// ---------------------------------------------------------------------------
// Kernel 4: output projection (8192 x 1024) * (1024 x 1024) + bias -> fp32
// ---------------------------------------------------------------------------
__launch_bounds__(256)
__global__ void k_gemm_proj(const unsigned short* __restrict__ Ab,
                            const unsigned short* __restrict__ Wpb,
                            const float* __restrict__ bp,
                            float* __restrict__ out) {
    __shared__ unsigned int   ldsA[2][128 * 20];
    __shared__ unsigned short ldsBt[2][128 * 34];

    const int tid  = threadIdx.x;
    const int wave = tid >> 5;
    const int lane = tid & 31;
    const int lrow = lane & 15;
    const int half = lane >> 4;
    const int wm = wave & 3, wn = wave >> 2;
    const int t0 = blockIdx.x * 128;
    const int n0 = blockIdx.y * 128;

    const unsigned short* Au = Ab;
    const unsigned int*   Wu = (const unsigned int*)Wpb;
    const unsigned ldsA_base0 = lds_off(&ldsA[0][0]);
    const unsigned ldsA_base1 = lds_off(&ldsA[1][0]);

    v8f acc[2][4];
#pragma unroll
    for (int mi = 0; mi < 2; mi++)
#pragma unroll
        for (int j = 0; j < 4; j++)
#pragma unroll
            for (int e = 0; e < 8; e++) acc[mi][j][e] = 0.0f;

    gemm_load_tile(Au, Wu, ldsA_base0, ldsBt[0], tid, t0, n0, 0);
    wait_async0();
    __syncthreads();
    for (int kt = 0; kt < 32; kt++) {
        int buf = kt & 1;
        if (kt + 1 < 32)
            gemm_load_tile(Au, Wu, buf ? ldsA_base0 : ldsA_base1,
                           ldsBt[buf ^ 1], tid, t0, n0, (kt + 1) * 32);
        gemm_compute(ldsA[buf], ldsBt[buf], wm, wn, lrow, half, acc);
        wait_async0();
        __syncthreads();
    }

#pragma unroll
    for (int mi = 0; mi < 2; mi++)
#pragma unroll
        for (int j = 0; j < 4; j++) {
            int n = n0 + wn * 64 + j * 16 + lrow;
            float bias = bp[n];
#pragma unroll
            for (int r = 0; r < 8; r++) {
                int tg = t0 + wm * 32 + mi * 16 + r + 8 * half;
                out[(size_t)tg * 1024 + n] = acc[mi][j][r] + bias;
            }
        }
}

// ---------------------------------------------------------------------------
extern "C" void kernel_launch(void* const* d_in, const int* in_sizes, int n_in,
                              void* d_out, int out_size, void* d_ws, size_t ws_size,
                              hipStream_t stream) {
    const float* x  = (const float*)d_in[0];
    const float* Wq = (const float*)d_in[1];
    const float* Wk = (const float*)d_in[2];
    const float* Wv = (const float*)d_in[3];
    const float* Wp = (const float*)d_in[4];
    const float* bp = (const float*)d_in[5];
    float* out = (float*)d_out;

    char* ws = (char*)d_ws;
    size_t off = 0;
    auto alloc = [&](size_t bytes) -> void* {
        void* p = ws + off;
        off += (bytes + 255) & ~(size_t)255;
        return p;
    };
    const size_t NXB  = (size_t)B_ * T_ * C_;            // 8,388,608
    const size_t NQKV = (size_t)B_ * H_ * T_ * HD_;      // 8,388,608
    unsigned short* xb    = (unsigned short*)alloc(NXB * 2);
    unsigned short* wqkvb = (unsigned short*)alloc((size_t)3 * C_ * C_ * 2);
    unsigned short* wpb   = (unsigned short*)alloc((size_t)C_ * C_ * 2);
    unsigned short* qb    = (unsigned short*)alloc(NQKV * 2);
    unsigned short* kbuf  = (unsigned short*)alloc(NQKV * 2);
    unsigned short* vbuf  = (unsigned short*)alloc(NQKV * 2);
    unsigned short* attO  = (unsigned short*)alloc(NXB * 2);
    (void)ws_size; (void)in_sizes; (void)n_in; (void)out_size;

    k_cvt_bf16<<<(int)((NXB / 2 + 255) / 256), 256, 0, stream>>>(
        x, (unsigned int*)xb, (int)(NXB / 2));
    k_pack_qkv<<<(C_ * C_ / 2 + 255) / 256, 256, 0, stream>>>(
        Wq, Wk, Wv, (unsigned int*)wqkvb);
    k_cvt_bf16<<<(C_ * C_ / 2 + 255) / 256, 256, 0, stream>>>(
        Wp, (unsigned int*)wpb, C_ * C_ / 2);

    k_gemm_qkv<<<dim3((B_ * T_) / 128, C_ / 128, 3), 256, 0, stream>>>(
        xb, wqkvb, qb, kbuf, vbuf);

    k_attn<<<dim3(T_ / 64, H_, B_), 128, 0, stream>>>(qb, kbuf, vbuf, attO);

    k_gemm_proj<<<dim3((B_ * T_) / 128, C_ / 128, 1), 256, 0, stream>>>(
        attO, wpb, bp, out);
}
